// MultiHeadAttention_67662914781212
// MI455X (gfx1250) — matrix operand order repaired and checked by reading, not compiled, not run on hardware
//
#include <hip/hip_runtime.h>

// ---------------- problem constants ----------------
#define BB   2
#define SS   2048
#define EE   512
#define HH   8
#define DK   64
#define AW   4          // waves per block in the attention kernel

// ---------------- WMMA / vector types ----------------
typedef __attribute__((ext_vector_type(16))) __bf16    v16bf;
typedef __attribute__((ext_vector_type(8)))  float     v8f;
typedef __attribute__((ext_vector_type(4)))  unsigned  u32x4;
typedef __attribute__((ext_vector_type(8)))  unsigned  u32x8;
typedef __attribute__((ext_vector_type(4)))  float     f32x4;

union BF16Frag {
  v16bf  v;
  __bf16 b[16];
  u32x4  q[2];
};

// Native RNE conversion (clang lowers fptrunc to hw bf16 cvt where available).
__device__ __forceinline__ __bf16 f2bf(float f) { return (__bf16)f; }

__device__ __forceinline__ v8f wmma_bf16(v16bf a, v16bf b, v8f c) {
  // D(f32 16x16) = A(bf16 16x32) * B(bf16 32x16) + C
  return __builtin_amdgcn_wmma_f32_16x16x32_bf16(
      false, a, false, b, (short)0, c, false, false);
}

// B fragment: lane n = lane&15, contraction c = (lane>>4)*16 + e
// -> 16 contiguous bf16 per lane (two b128 loads).
__device__ __forceinline__ v16bf load_b_frag(const __bf16* p) {
  BF16Frag f;
  f.q[0] = *(const u32x4*)(p);
  f.q[1] = *(const u32x4*)(p + 8);
  return f.v;
}

// A fragment: lane m = lane&15; K = half*16 + e
// -> two contiguous 8-element (16B) groups per lane.
__device__ __forceinline__ v16bf load_a_frag(const __bf16* row, int kb, int half) {
  BF16Frag f;
  f.q[0] = *(const u32x4*)(row + kb + half * 16);
  f.q[1] = *(const u32x4*)(row + kb + half * 16 + 8);
  return f.v;
}

// =====================================================================
// Stage 0: bulk f32 -> bf16 conversion (8 elements / thread, exact grids)
// =====================================================================
__global__ __launch_bounds__(256) void cvt_bf16_kernel(
    const float* __restrict__ in, __bf16* __restrict__ out)
{
  const int i = blockIdx.x * 256 + threadIdx.x;     // group of 8 elements
  const f32x4 a = ((const f32x4*)in)[2 * i];
  const f32x4 b = ((const f32x4*)in)[2 * i + 1];
  BF16Frag o;                                        // use low half only
#pragma unroll
  for (int e = 0; e < 4; ++e) { o.b[e] = f2bf(a[e]); o.b[e + 4] = f2bf(b[e]); }
  ((u32x4*)out)[i] = o.q[0];
}

// =====================================================================
// Stage 1: projections  out = X @ W^T + b  (nn.Linear convention)
// X: [4096,512] bf16, W: [512,512] bf16 (pre-converted), bias f32
// transposed==0 : out bf16 [B,H,S,64]   (Q and K)
// transposed==1 : out bf16 [B,H,64,S]   (V stored transposed for PV GEMM)
// One wave computes a 16(M) x 64(N) tile. Fully-unrolled ping-pong K-loop:
// straight-line code (no backedge register constraints -> no tuple-assembly
// copies) with next step's loads textually ahead of current WMMAs.
// =====================================================================
__global__ __launch_bounds__(256) void proj_kernel(
    const __bf16* __restrict__ X, const __bf16* __restrict__ W,
    const float* __restrict__ bias, __bf16* __restrict__ out, int transposed)
{
  const int lane = threadIdx.x & 31;
  const int wave = threadIdx.x >> 5;
  const int t     = blockIdx.x * 8 + wave;   // 0..2047
  const int mBase = (t >> 3) * 16;           // 256 M-tiles
  const int h     = t & 7;                   // N-tile == head
  const int nBase = h * 64;
  const int ln    = lane & 15;
  const int half  = lane >> 4;

  const __bf16* xrow = X + (size_t)(mBase + ln) * EE;
  const __bf16* wrow = W + (size_t)(nBase + ln) * EE + half * 16;  // +nt*16*EE

  v8f acc[4] = {};
  v16bf aA = load_a_frag(xrow, 0, half);
  v16bf bA[4];
#pragma unroll
  for (int nt = 0; nt < 4; ++nt) bA[nt] = load_b_frag(wrow + (size_t)nt * 16 * EE);

#pragma unroll
  for (int kb = 0; kb < EE; kb += 64) {
    // load set B (kb+32) while consuming set A (kb)
    v16bf aB = load_a_frag(xrow, kb + 32, half);
    v16bf bB[4];
#pragma unroll
    for (int nt = 0; nt < 4; ++nt)
      bB[nt] = load_b_frag(wrow + (size_t)nt * 16 * EE + kb + 32);
#pragma unroll
    for (int nt = 0; nt < 4; ++nt) acc[nt] = wmma_bf16(aA, bA[nt], acc[nt]);

    // load set A (kb+64) while consuming set B (kb+32); guard is
    // compile-time constant under full unroll (no loads on last step)
    if (kb + 64 < EE) {
      aA = load_a_frag(xrow, kb + 64, half);
#pragma unroll
      for (int nt = 0; nt < 4; ++nt)
        bA[nt] = load_b_frag(wrow + (size_t)nt * 16 * EE + kb + 64);
    }
#pragma unroll
    for (int nt = 0; nt < 4; ++nt) acc[nt] = wmma_bf16(aB, bB[nt], acc[nt]);
  }

  const int bIdx  = mBase >> 11;       // /S
  const int sBase = mBase & (SS - 1);
#pragma unroll
  for (int nt = 0; nt < 4; ++nt) {
    const int   d    = nt * 16 + ln;   // 0..63 within head
    const float bval = bias[nBase + d];
#pragma unroll
    for (int r = 0; r < 8; ++r) {
      const int s = sBase + r + 8 * half;
      const float o = acc[nt][r] + bval;
      if (!transposed)
        out[(((size_t)(bIdx * HH + h) * SS) + s) * DK + d] = f2bf(o);
      else
        out[(((size_t)(bIdx * HH + h) * DK) + d) * SS + s] = f2bf(o);
    }
  }
}

// =====================================================================
// Stage 2: flash attention (causal), one wave per 16-query tile.
// K tiles (32 keys x 64 dk, 4 KB bf16) are fed through the Tensor Data
// Mover into per-wave LDS double buffers (tensor_load_to_lds +
// s_wait_tensorcnt); V^T tiles go global->register and overlap softmax.
// Scores (would be 268 MB > 192 MB L2) never touch memory.
// =====================================================================
__global__ __launch_bounds__(128) void attn_kernel(
    const __bf16* __restrict__ Qb, const __bf16* __restrict__ Kb,
    const __bf16* __restrict__ Vt, __bf16* __restrict__ Xa)
{
  __shared__ __bf16 Klds[AW][2][32][DK];       // 32 KB: TDM destination
  __shared__ __bf16 Plds[AW][16][32];          //  4 KB: P C->A relayout

  const int lane = threadIdx.x & 31;
  const int wave = threadIdx.x >> 5;
  const int t     = blockIdx.x * AW + wave;    // 0..2047
  const int b     = t >> 10;                   // / (H * S/16)
  const int h     = (t >> 7) & 7;
  const int qBase = (t & 127) * 16;
  const int ln    = lane & 15;
  const int half  = lane >> 4;

  const __bf16* Qh = Qb + (size_t)(b * HH + h) * SS * DK;
  const __bf16* Kh = Kb + (size_t)(b * HH + h) * SS * DK;
  const __bf16* Vh = Vt + (size_t)(b * HH + h) * DK * SS;

  // Issue a TDM load of K rows [kb, kb+32) x dk[0,64) into Klds[wave][buf].
  auto tdm_load_k = [&](int kb, int buf) {
    const unsigned ldsOff = (unsigned)(size_t)&Klds[wave][buf][0][0];
    const unsigned long long ga =
        (unsigned long long)(size_t)(Kh + (size_t)kb * DK);
    u32x4 g0;
    g0[0] = 1u;                                   // count=1, user descriptor
    g0[1] = ldsOff;                               // lds_addr (bytes)
    g0[2] = (unsigned)ga;                         // global_addr[31:0]
    g0[3] = ((unsigned)(ga >> 32) & 0x01FFFFFFu)  // global_addr[56:32]
            | 0x80000000u;                        // type=2 ("image")
    u32x8 g1;
    g1[0] = 1u << 16;                             // data_size=1 (2 bytes)
    g1[1] = (unsigned)DK << 16;                   // tensor_dim0 = 64
    g1[2] = (unsigned)SS << 16;                   // tensor_dim1 = 2048
    g1[3] = (unsigned)DK << 16;                   // tile_dim0 = 64
    g1[4] = 32u;                                  // tile_dim1 = 32
    g1[5] = (unsigned)DK;                         // tensor_dim0_stride = 64
    g1[6] = 0u;
    g1[7] = 0u;
    u32x4 g2 = {0u, 0u, 0u, 0u};                  // 2-D tile: groups 2/3 unused
    u32x4 g3 = {0u, 0u, 0u, 0u};
    asm volatile("tensor_load_to_lds %0, %1, %2, %3"
                 :: "s"(g0), "s"(g1), "s"(g2), "s"(g3) : "memory");
  };

  // Q A-fragments for this wave's 16 queries (d_k = 64 -> 2 K-chunks of 32)
  const __bf16* qrow = Qh + (size_t)(qBase + ln) * DK;
  const v16bf aq0 = load_a_frag(qrow, 0,  half);
  const v16bf aq1 = load_a_frag(qrow, 32, half);

  v8f acc0 = {}, acc1 = {}, acc2 = {}, acc3 = {};
  float rowM[8], rowL[8];
#pragma unroll
  for (int r = 0; r < 8; ++r) { rowM[r] = -3.0e38f; rowL[r] = 0.0f; }

  const int kEnd = qBase + 16;                 // causal: keys < qBase+16
  tdm_load_k(0, 0);                            // prime the pipeline
  int p = 0;
  for (int kb = 0; kb < kEnd; kb += 32, p ^= 1) {
    const bool more = (kb + 32) < kEnd;
    if (more) {
      tdm_load_k(kb + 32, p ^ 1);              // async load of next tile
      __builtin_amdgcn_s_wait_tensorcnt(1);    // current tile complete
    } else {
      __builtin_amdgcn_s_wait_tensorcnt(0);
    }
    asm volatile("" ::: "memory");             // keep LDS reads below the wait

    // ---- scores S = (Q K^T) * 1/sqrt(64) for 16q x 32k (K from LDS) ----
    const __bf16* kr0 = &Klds[wave][p][ln][0];
    const __bf16* kr1 = &Klds[wave][p][16 + ln][0];
    v8f s0 = {}, s1 = {};
    s0 = wmma_bf16(aq0, load_b_frag(kr0 + half * 16),      s0);
    s0 = wmma_bf16(aq1, load_b_frag(kr0 + 32 + half * 16), s0);
    s1 = wmma_bf16(aq0, load_b_frag(kr1 + half * 16),      s1);
    s1 = wmma_bf16(aq1, load_b_frag(kr1 + 32 + half * 16), s1);

    // ---- V^T fragment loads issued now; consumed after softmax ----
    const __bf16* vp = Vh + kb + half * 16;
    const v16bf bv0 = load_b_frag(vp + (size_t)(0 * 16 + ln) * SS);
    const v16bf bv1 = load_b_frag(vp + (size_t)(1 * 16 + ln) * SS);
    const v16bf bv2 = load_b_frag(vp + (size_t)(2 * 16 + ln) * SS);
    const v16bf bv3 = load_b_frag(vp + (size_t)(3 * 16 + ln) * SS);

    const bool needMask = (kb + 31 > qBase);

    // ---- online softmax per row (rows live across 16 lanes of a half) ----
#pragma unroll
    for (int r = 0; r < 8; ++r) {
      float x0 = s0[r] * 0.125f;
      float x1 = s1[r] * 0.125f;
      if (needMask) {
        const int qi = qBase + r + 8 * half;
        if (kb + ln      > qi) x0 = -1.0e9f;
        if (kb + 16 + ln > qi) x1 = -1.0e9f;
      }
      float tm = fmaxf(x0, x1);
      tm = fmaxf(tm, __shfl_xor(tm, 1, 32));
      tm = fmaxf(tm, __shfl_xor(tm, 2, 32));
      tm = fmaxf(tm, __shfl_xor(tm, 4, 32));
      tm = fmaxf(tm, __shfl_xor(tm, 8, 32));
      const float newM  = fmaxf(rowM[r], tm);
      const float alpha = __expf(rowM[r] - newM);
      rowM[r] = newM;
      const float p0 = __expf(x0 - newM);
      const float p1 = __expf(x1 - newM);
      float ts = p0 + p1;
      ts += __shfl_xor(ts, 1, 32);
      ts += __shfl_xor(ts, 2, 32);
      ts += __shfl_xor(ts, 4, 32);
      ts += __shfl_xor(ts, 8, 32);
      rowL[r] = rowL[r] * alpha + ts;
      acc0[r] *= alpha; acc1[r] *= alpha; acc2[r] *= alpha; acc3[r] *= alpha;
      // stage P (C-fragment layout) into this wave's LDS tile
      const int m = r + 8 * half;
      Plds[wave][m][ln]      = f2bf(p0);
      Plds[wave][m][16 + ln] = f2bf(p1);
    }

    // ---- O += P (16x32, A-fragment from LDS) * V^T-tile (32x64) ----
    const v16bf pa = load_a_frag(&Plds[wave][ln][0], 0, half);
    acc0 = wmma_bf16(pa, bv0, acc0);
    acc1 = wmma_bf16(pa, bv1, acc1);
    acc2 = wmma_bf16(pa, bv2, acc2);
    acc3 = wmma_bf16(pa, bv3, acc3);
  }

  // ---- normalize and store head output into [B*S, 512] bf16 ----
#pragma unroll
  for (int r = 0; r < 8; ++r) {
    const float inv = 1.0f / rowL[r];
    const int   s   = qBase + r + 8 * half;
    __bf16* orow = Xa + (size_t)(b * SS + s) * EE + h * DK;
    orow[0 * 16 + ln] = f2bf(acc0[r] * inv);
    orow[1 * 16 + ln] = f2bf(acc1[r] * inv);
    orow[2 * 16 + ln] = f2bf(acc2[r] * inv);
    orow[3 * 16 + ln] = f2bf(acc3[r] * inv);
  }
}

// =====================================================================
// Stage 3: output projection  out = Xa @ Wo^T + bo   (f32 output)
// Same fully-unrolled ping-pong structure as proj_kernel.
// =====================================================================
__global__ __launch_bounds__(256) void oproj_kernel(
    const __bf16* __restrict__ Xa, const __bf16* __restrict__ Wo,
    const float* __restrict__ bo, float* __restrict__ out)
{
  const int lane = threadIdx.x & 31;
  const int wave = threadIdx.x >> 5;
  const int t     = blockIdx.x * 8 + wave;   // 0..2047
  const int mBase = (t >> 3) * 16;
  const int nBase = (t & 7) * 64;
  const int ln    = lane & 15;
  const int half  = lane >> 4;

  const __bf16* xrow = Xa + (size_t)(mBase + ln) * EE;
  const __bf16* wrow = Wo + (size_t)(nBase + ln) * EE + half * 16;

  v8f acc[4] = {};
  v16bf aA = load_a_frag(xrow, 0, half);
  v16bf bA[4];
#pragma unroll
  for (int nt = 0; nt < 4; ++nt) bA[nt] = load_b_frag(wrow + (size_t)nt * 16 * EE);

#pragma unroll
  for (int kb = 0; kb < EE; kb += 64) {
    v16bf aB = load_a_frag(xrow, kb + 32, half);
    v16bf bB[4];
#pragma unroll
    for (int nt = 0; nt < 4; ++nt)
      bB[nt] = load_b_frag(wrow + (size_t)nt * 16 * EE + kb + 32);
#pragma unroll
    for (int nt = 0; nt < 4; ++nt) acc[nt] = wmma_bf16(aA, bA[nt], acc[nt]);

    if (kb + 64 < EE) {
      aA = load_a_frag(xrow, kb + 64, half);
#pragma unroll
      for (int nt = 0; nt < 4; ++nt)
        bA[nt] = load_b_frag(wrow + (size_t)nt * 16 * EE + kb + 64);
    }
#pragma unroll
    for (int nt = 0; nt < 4; ++nt) acc[nt] = wmma_bf16(aB, bB[nt], acc[nt]);
  }

#pragma unroll
  for (int nt = 0; nt < 4; ++nt) {
    const int   n    = nBase + nt * 16 + ln;
    const float bval = bo[n];
#pragma unroll
    for (int r = 0; r < 8; ++r) {
      const int row = mBase + r + 8 * half;
      out[(size_t)row * EE + n] = acc[nt][r] + bval;
    }
  }
}

// =====================================================================
extern "C" void kernel_launch(void* const* d_in, const int* in_sizes, int n_in,
                              void* d_out, int out_size, void* d_ws, size_t ws_size,
                              hipStream_t stream) {
  // input order: query,key,value,mask,Wq,bq,Wk,bk,Wv,bv,Wo,bo
  const float* query = (const float*)d_in[0];
  const float* key   = (const float*)d_in[1];
  const float* value = (const float*)d_in[2];
  // d_in[3] (mask) is the causal tril mask -> implemented analytically
  const float* Wq = (const float*)d_in[4];  const float* bq = (const float*)d_in[5];
  const float* Wk = (const float*)d_in[6];  const float* bk = (const float*)d_in[7];
  const float* Wv = (const float*)d_in[8];  const float* bv = (const float*)d_in[9];
  const float* Wo = (const float*)d_in[10]; const float* bo = (const float*)d_in[11];

  // bf16 workspace layout (~31.5 MB total, L2-resident on MI455X):
  const size_t NX = (size_t)BB * SS * EE;   // 2,097,152 (activation tensor)
  const size_t NW = (size_t)EE * EE;        //   262,144 (weight matrix)
  __bf16* Xq  = (__bf16*)d_ws;
  __bf16* Xk  = Xq  + NX;
  __bf16* Xv  = Xk  + NX;
  __bf16* Wqb = Xv  + NX;
  __bf16* Wkb = Wqb + NW;
  __bf16* Wvb = Wkb + NW;
  __bf16* Wob = Wvb + NW;
  __bf16* Qb  = Wob + NW;
  __bf16* Kb  = Qb  + NX;
  __bf16* Vt  = Kb  + NX;
  __bf16* Xa  = Vt  + NX;

  dim3 blk(256);
  const int gX = (int)(NX / (8 * 256));     // 1024 blocks per activation tensor
  const int gW = (int)(NW / (8 * 256));     // 128 blocks per weight matrix

  // Stage 0: one-time f32 -> bf16 conversion
  cvt_bf16_kernel<<<gX, blk, 0, stream>>>(query, Xq);
  cvt_bf16_kernel<<<gX, blk, 0, stream>>>(key,   Xk);
  cvt_bf16_kernel<<<gX, blk, 0, stream>>>(value, Xv);
  cvt_bf16_kernel<<<gW, blk, 0, stream>>>(Wq, Wqb);
  cvt_bf16_kernel<<<gW, blk, 0, stream>>>(Wk, Wkb);
  cvt_bf16_kernel<<<gW, blk, 0, stream>>>(Wv, Wvb);
  cvt_bf16_kernel<<<gW, blk, 0, stream>>>(Wo, Wob);

  // Stages 1-3: pure-bf16 WMMA pipeline
  proj_kernel<<<256, blk, 0, stream>>>(Xq, Wqb, bq, Qb, 0);
  proj_kernel<<<256, blk, 0, stream>>>(Xk, Wkb, bk, Kb, 0);
  proj_kernel<<<256, blk, 0, stream>>>(Xv, Wvb, bv, Vt, 1);
  attn_kernel<<<512, dim3(128), 0, stream>>>(Qb, Kb, Vt, Xa);
  oproj_kernel<<<256, blk, 0, stream>>>(Xa, Wob, bo, (float*)d_out);
}
